// DecoderLSTM_1838246003177
// MI455X (gfx1250) — matrix operand order repaired and checked by reading, not compiled, not run on hardware
//
#include <hip/hip_runtime.h>
#include <hip/hip_bf16.h>

// ---------------------------------------------------------------------------
// DecoderLSTM on gfx1250: 12 sequential LSTM steps, each dominated by
// gates = h @ W_hh^T ([32768,512] x [512,2048]) via v_wmma_f32_16x16x32_bf16.
// W_hh K-chunks are staged into LDS with global_load_async_to_lds_b128
// (double-buffered, ASYNCcnt-tracked) so each block loads B once instead of
// once per wave; WMMA B fragments are then fed from LDS via ds_load_b128.
// ---------------------------------------------------------------------------

typedef __bf16  bf16_t;
typedef __bf16  v16bf __attribute__((ext_vector_type(16)));
typedef __bf16  v8bf  __attribute__((ext_vector_type(8)));
typedef float   v8f   __attribute__((ext_vector_type(8)));

static constexpr int NB = 32768;  // batch
static constexpr int HH = 512;    // hidden
static constexpr int OO = 2;      // output size
static constexpr int TT = 12;     // pred_len

static constexpr int LDS_STRIDE = 40;  // bf16 elems per staged row (32 + 8 pad = 80B)

// ---------------------------------------------------------------------------
// Init: h0 = bf16(context), c0 = 0, W_hh -> bf16, bias = b_ih + b_hh
// ---------------------------------------------------------------------------
__global__ __launch_bounds__(256)
void lstm_init_kernel(const float* __restrict__ context,
                      const float* __restrict__ Whh_f,
                      const float* __restrict__ b_ih,
                      const float* __restrict__ b_hh,
                      bf16_t* __restrict__ h0,
                      float*  __restrict__ c0,
                      bf16_t* __restrict__ Whh_bf,
                      float*  __restrict__ bias)
{
    const int tid  = blockIdx.x * blockDim.x + threadIdx.x;
    const int nthr = gridDim.x * blockDim.x;

    for (int i = tid; i < NB * HH; i += nthr) {
        h0[i] = (bf16_t)context[i];
        c0[i] = 0.0f;
    }
    for (int i = tid; i < 4 * HH * HH; i += nthr) {
        Whh_bf[i] = (bf16_t)Whh_f[i];
    }
    for (int i = tid; i < 4 * HH; i += nthr) {
        bias[i] = b_ih[i] + b_hh[i];
    }
}

// ---------------------------------------------------------------------------
// One LSTM step.
// Block: 256 threads = 8 waves. Tile: 128 batch rows x 32 hidden columns,
// computing all 4 gate tiles (cols j, H+j, 2H+j, 3H+j) so the cell update
// fuses in-register. Wave w owns rows mBase + w*16 .. +15.
// B (W_hh chunk, 128 rows x 32 K) is async-staged into LDS once per block.
// ---------------------------------------------------------------------------
__global__ __launch_bounds__(256)
void lstm_step_kernel(const bf16_t* __restrict__ h_in,   // [N,H] bf16
                      bf16_t*       __restrict__ h_out,  // [N,H] bf16
                      float*        __restrict__ c,      // [N,H] f32, in-place
                      const bf16_t* __restrict__ Whh,    // [4H,H] bf16 row-major
                      const float*  __restrict__ Wih,    // [4H,2] f32
                      const float*  __restrict__ bias,   // [4H]  f32 (b_ih+b_hh)
                      const float*  __restrict__ x0,     // x for n=0
                      int x_stride)                      // row stride of x
{
    __shared__ __align__(16) bf16_t smemB[2][128 * LDS_STRIDE];  // 2 x 10 KB

    const int lane  = threadIdx.x & 31;
    const int wave  = threadIdx.x >> 5;
    const int lm    = lane & 15;   // 0..15 : M for A, N for B/C
    const int khalf = lane >> 4;   // 0/1   : K-half selector

    const int mBase = blockIdx.x * 128 + wave * 16;
    const int jBase = blockIdx.y * 32;

    // Stage one 128x32 W_hh K-chunk into LDS buffer `buf` with the async
    // copy engine: 512 b128 pieces, 2 per thread, tracked by ASYNCcnt.
    auto stage = [&](int buf, int k0) {
#pragma unroll
        for (int i = 0; i < 2; ++i) {
            const int pid   = threadIdx.x + i * 256;  // 0..511
            const int row   = pid >> 2;               // 0..127 local weight row
            const int piece = pid & 3;                // 16B piece within 64B row
            const int g  = row >> 5;
            const int jl = row & 31;
            const unsigned voff =
                (unsigned)(((g * HH + jBase + jl) * HH + k0) * 2 + piece * 16);
            const unsigned loff =
                (unsigned)(size_t)&smemB[buf][0] +
                (unsigned)(row * (LDS_STRIDE * 2) + piece * 16);
            asm volatile("global_load_async_to_lds_b128 %0, %1, %2"
                         :: "v"(loff), "v"(voff), "s"(Whh)
                         : "memory");
        }
    };

    v8f acc[4][2];
#pragma unroll
    for (int g = 0; g < 4; ++g)
#pragma unroll
        for (int s = 0; s < 2; ++s)
            acc[g][s] = {};

    // A fragment source: row mBase+lm of h_in.
    const bf16_t* aPtr = h_in + (size_t)(mBase + lm) * HH;

    stage(0, 0);

    for (int kk = 0; kk < HH / 32; ++kk) {
        const int k0 = kk * 32;

        if (kk + 1 < HH / 32) {
            stage((kk + 1) & 1, k0 + 32);
            // 2 outstanding ops belong to chunk kk+1; chunk kk is complete.
            asm volatile("s_wait_asynccnt 0x2" ::: "memory");
        } else {
            asm volatile("s_wait_asynccnt 0x0" ::: "memory");
        }
        __syncthreads();  // chunk kk visible to all waves

        // --- A fragment (16x32 bf16): lanes 0-15 hold K {0..7,16..23},
        //     lanes 16-31 hold K {8..15,24..31} of this K-chunk.
        v8bf alo = *(const v8bf*)(aPtr + k0 + khalf * 8);
        v8bf ahi = *(const v8bf*)(aPtr + k0 + 16 + khalf * 8);
        v16bf a;
#pragma unroll
        for (int i = 0; i < 8; ++i) { a[i] = alo[i]; a[8 + i] = ahi[i]; }

        const bf16_t* bbase = &smemB[kk & 1][0];
#pragma unroll
        for (int g = 0; g < 4; ++g) {
#pragma unroll
            for (int s = 0; s < 2; ++s) {
                // --- B fragment (32x16 bf16) from LDS: lane = column n,
                //     K packed along VGPRs; khalf selects K {0..7,16..23} vs {8..15,24..31}.
                const int colLocal = g * 32 + s * 16 + lm;
                const bf16_t* p = bbase + colLocal * LDS_STRIDE + khalf * 8;
                v8bf b0 = *(const v8bf*)(p);
                v8bf b1 = *(const v8bf*)(p + 16);
                v16bf b;
#pragma unroll
                for (int i = 0; i < 8; ++i) { b[i] = b0[i]; b[8 + i] = b1[i]; }

                acc[g][s] = __builtin_amdgcn_wmma_f32_16x16x32_bf16(
                    /*neg_a=*/false, a, /*neg_b=*/false, b,
                    /*c_mod=*/(short)0, acc[g][s],
                    /*reuse_a=*/false, /*reuse_b=*/false);
            }
        }
        __syncthreads();  // all waves done reading before buffer is re-staged
    }

    // ------------------------------------------------------------------
    // Cell update. C/D layout: acc[r] on lane L holds element
    // (M = r + (L>=16 ? 8 : 0), N = L&15).
    // ------------------------------------------------------------------
#pragma unroll
    for (int s = 0; s < 2; ++s) {
        const int j = jBase + s * 16 + lm;
        const float bi = bias[0 * HH + j];
        const float bf = bias[1 * HH + j];
        const float bg = bias[2 * HH + j];
        const float bo = bias[3 * HH + j];
        const float wi0 = Wih[(0 * HH + j) * OO + 0], wi1 = Wih[(0 * HH + j) * OO + 1];
        const float wf0 = Wih[(1 * HH + j) * OO + 0], wf1 = Wih[(1 * HH + j) * OO + 1];
        const float wg0 = Wih[(2 * HH + j) * OO + 0], wg1 = Wih[(2 * HH + j) * OO + 1];
        const float wo0 = Wih[(3 * HH + j) * OO + 0], wo1 = Wih[(3 * HH + j) * OO + 1];

#pragma unroll
        for (int r = 0; r < 8; ++r) {
            const int m = mBase + r + khalf * 8;
            const float xv0 = x0[(size_t)m * x_stride + 0];
            const float xv1 = x0[(size_t)m * x_stride + 1];

            float gi = acc[0][s][r] + bi + xv0 * wi0 + xv1 * wi1;
            float gf = acc[1][s][r] + bf + xv0 * wf0 + xv1 * wf1;
            float gg = acc[2][s][r] + bg + xv0 * wg0 + xv1 * wg1;
            float go = acc[3][s][r] + bo + xv0 * wo0 + xv1 * wo1;

            gi = 1.0f / (1.0f + __expf(-gi));
            gf = 1.0f / (1.0f + __expf(-gf));
            go = 1.0f / (1.0f + __expf(-go));
            gg = tanhf(gg);

            const size_t off = (size_t)m * HH + j;
            const float cv = gf * c[off] + gi * gg;
            c[off]     = cv;
            h_out[off] = (bf16_t)(go * tanhf(cv));
        }
    }
}

// ---------------------------------------------------------------------------
// pred = h @ W_fc^T + b_fc, written straight into d_out[n, t, :].
// ---------------------------------------------------------------------------
__global__ __launch_bounds__(256)
void lstm_pred_kernel(const bf16_t* __restrict__ h,     // [N,H] bf16
                      const float*  __restrict__ Wfc,   // [2,H] f32
                      const float*  __restrict__ bfc,   // [2]
                      float*        __restrict__ out)   // &d_out[0, t, 0]
{
    const int n = blockIdx.x * blockDim.x + threadIdx.x;
    if (n >= NB) return;
    const bf16_t* hp = h + (size_t)n * HH;
    float s0 = 0.0f, s1 = 0.0f;
    for (int k = 0; k < HH; k += 8) {
        v8bf hv = *(const v8bf*)(hp + k);
#pragma unroll
        for (int i = 0; i < 8; ++i) {
            const float hvf = (float)hv[i];
            s0 += hvf * Wfc[0 * HH + k + i];
            s1 += hvf * Wfc[1 * HH + k + i];
        }
    }
    out[(size_t)n * (TT * OO) + 0] = s0 + bfc[0];
    out[(size_t)n * (TT * OO) + 1] = s1 + bfc[1];
}

// ---------------------------------------------------------------------------
extern "C" void kernel_launch(void* const* d_in, const int* in_sizes, int n_in,
                              void* d_out, int out_size, void* d_ws, size_t ws_size,
                              hipStream_t stream)
{
    const float* context   = (const float*)d_in[0];  // [N,H]
    const float* start_pos = (const float*)d_in[1];  // [N,O]
    const float* W_ih      = (const float*)d_in[2];  // [4H,O]
    const float* W_hh      = (const float*)d_in[3];  // [4H,H]
    const float* b_ih      = (const float*)d_in[4];  // [4H]
    const float* b_hh      = (const float*)d_in[5];  // [4H]
    const float* W_fc      = (const float*)d_in[6];  // [O,H]
    const float* b_fc      = (const float*)d_in[7];  // [O]
    float* out             = (float*)d_out;          // [N,T,O]

    // Workspace carve-up (~130 MB)
    char* ws = (char*)d_ws;
    bf16_t* hA     = (bf16_t*)ws; ws += (size_t)NB * HH * sizeof(bf16_t);    // 32 MB
    bf16_t* hB     = (bf16_t*)ws; ws += (size_t)NB * HH * sizeof(bf16_t);    // 32 MB
    float*  c      = (float*) ws; ws += (size_t)NB * HH * sizeof(float);     // 64 MB
    bf16_t* Whh_bf = (bf16_t*)ws; ws += (size_t)4 * HH * HH * sizeof(bf16_t);// 2 MB
    float*  bias   = (float*) ws; ws += (size_t)4 * HH * sizeof(float);

    lstm_init_kernel<<<2048, 256, 0, stream>>>(context, W_hh, b_ih, b_hh,
                                               hA, c, Whh_bf, bias);

    dim3 gGates(NB / 128, HH / 32);   // 256 x 16 blocks
    bf16_t* hin = hA;
    bf16_t* hout = hB;
    for (int t = 0; t < TT; ++t) {
        const float* x = (t == 0) ? start_pos : (out + (size_t)(t - 1) * OO);
        const int    xs = (t == 0) ? OO : TT * OO;
        lstm_step_kernel<<<gGates, 256, 0, stream>>>(hin, hout, c, Whh_bf,
                                                     W_ih, bias, x, xs);
        lstm_pred_kernel<<<NB / 256, 256, 0, stream>>>(hout, W_fc, b_fc,
                                                       out + (size_t)t * OO);
        bf16_t* tmp = hin; hin = hout; hout = tmp;
    }
}
